// HybridQLSTM_65481071407217
// MI455X (gfx1250) — matrix operand order repaired and checked by reading, not compiled, not run on hardware
//
#include <hip/hip_runtime.h>
#include <hip/hip_bf16.h>

// ---------------------------------------------------------------------------
// LSTM on MI455X (gfx1250), wave32 + WMMA bf16.
//
//   T=1024, B=64, D=256, H=256.
//   Phase 1: convert W -> bf16                      (elementwise)
//   Phase 2: G[t,gate] = x_t @ Wx.T + b  (fp32)     (parallel WMMA GEMM)
//   Phase 3: persistent recurrent kernel, 16 blocks x 4 waves (1 wave/SIMD):
//            each wave owns one 16x16 (brow,hcol) tile for all 4 gates,
//            Wh fragments cached in VGPRs (256 VGPRs), c-state in VGPRs,
//            G software-pipelined one step ahead (NT loads),
//            1 grid barrier per step (cluster-barrier fast path + atomics),
//            double-buffered h in global (L2-resident, 32 KB).
//
// Workspace layout (needs ~270 MB):
//   [0)        Wbf  : 4*256*512 bf16              (1 MB)
//   [1MB)      G    : 1024*4*64*256 fp32          (256 MB)
//   [..)       hbuf : 2*64*256 bf16               (64 KB)
//   [..)       bar  : 2 x u32 barrier state
// ---------------------------------------------------------------------------

#define T_STEPS 1024
#define BSZ     64
#define DIN     256
#define HID     256
#define DH      512
#define NBLK_REC 16

typedef __attribute__((ext_vector_type(16))) __bf16 v16bf;
typedef __attribute__((ext_vector_type(8)))  float  v8f;

union FragB16 { v16bf v; uint4 q[2]; };   // 16 bf16 = 8 VGPRs
union AF16    { float4 f4[4]; float f[16]; };

static const size_t WBF_OFF  = 0;
static const size_t WBF_SZ   = (size_t)4 * HID * DH * 2;              // 1 MB
static const size_t G_OFF    = WBF_OFF + WBF_SZ;
static const size_t G_SZ     = (size_t)T_STEPS * 4 * BSZ * HID * 4;   // 256 MB
static const size_t HBUF_OFF = G_OFF + G_SZ;
static const size_t HBUF_SZ  = (size_t)2 * BSZ * HID * 2;             // 64 KB
static const size_t BAR_OFF  = HBUF_OFF + HBUF_SZ;

// ---- CDNA5 hardware transcendentals ---------------------------------------
__device__ __forceinline__ float tanh_hw(float x) {
  float r;
  asm("v_tanh_f32 %0, %1" : "=v"(r) : "v"(x));
  return r;
}
__device__ __forceinline__ float sigm_hw(float x) {
  // sigmoid(x) = 0.5*tanh(x/2) + 0.5  -> single trans op
  return __builtin_fmaf(0.5f, tanh_hw(0.5f * x), 0.5f);
}

// ---- Phase 1: weights fp32 -> bf16 ----------------------------------------
__global__ void cvt_weights(const float* __restrict__ Wf, const float* __restrict__ Wi,
                            const float* __restrict__ Wg, const float* __restrict__ Wo,
                            __hip_bfloat16* __restrict__ wbf) {
  const int per = HID * DH;                     // 131072
  int idx = blockIdx.x * blockDim.x + threadIdx.x;
  if (idx >= 4 * per) return;
  const float* src = (idx < per) ? Wf : (idx < 2*per) ? Wi : (idx < 3*per) ? Wg : Wo;
  wbf[idx] = __float2bfloat16(src[idx & (per - 1)]);
}

// ---- init h buffers + barrier state (ws is poisoned 0xAA, must clear) -----
__global__ void init_state(__hip_bfloat16* __restrict__ hbuf, unsigned* __restrict__ bar) {
  int idx = blockIdx.x * blockDim.x + threadIdx.x;
  if (idx < 2 * BSZ * HID) hbuf[idx] = __float2bfloat16(0.0f);
  if (idx < 2) bar[idx] = 0u;
}

// ---- Phase 2: G = x @ Wx.T + b  (bf16 WMMA, fp32 out) ----------------------
__global__ __launch_bounds__(256) void xproj(
    const float* __restrict__ x,                  // [T*B, D]
    const __hip_bfloat16* __restrict__ wbf,       // [4][H][DH]
    const float* __restrict__ bf_, const float* __restrict__ bi_,
    const float* __restrict__ bg_, const float* __restrict__ bo_,
    float* __restrict__ G) {                      // [T][4][B][H]
  const int lane = threadIdx.x & 31;
  const int wave = threadIdx.x >> 5;
  const int half = lane >> 4;
  const int l15  = lane & 15;

  long tid = (long)blockIdx.x * 8 + wave;        // tile id: (gate*4096 + mt)*16 + nt
  int  nt   = (int)(tid & 15);
  long rem  = tid >> 4;
  int  mt   = (int)(rem & 4095);
  int  gate = (int)(rem >> 12);

  const float* arow = x + (size_t)(mt * 16 + l15) * DIN;
  const __hip_bfloat16* brp = wbf + ((size_t)gate * HID + (size_t)(nt * 16 + l15)) * DH;

  const float* bias = (gate == 0) ? bf_ : (gate == 1) ? bi_ : (gate == 2) ? bg_ : bo_;
  float bv = bias[nt * 16 + l15];
  v8f acc;
  #pragma unroll
  for (int r = 0; r < 8; ++r) acc[r] = bv;

  #pragma unroll
  for (int kc = 0; kc < 8; ++kc) {
    // A fragment: 16 contiguous fp32 -> bf16 per lane at klo
    int klo = kc * 32 + half * 16;
    AF16 a;
    a.f4[0] = *(const float4*)(arow + klo);
    a.f4[1] = *(const float4*)(arow + klo + 4);
    a.f4[2] = *(const float4*)(arow + klo + 8);
    a.f4[3] = *(const float4*)(arow + klo + 12);
    v16bf av;
    #pragma unroll
    for (int j = 0; j < 16; ++j) av[j] = (__bf16)a.f[j];
    // B fragment: 16 contiguous bf16 per lane at k = kc*32 + half*16
    FragB16 b;
    int kb = kc * 32 + half * 16;
    b.q[0] = *(const uint4*)(brp + kb);
    b.q[1] = *(const uint4*)(brp + kb + 8);
    acc = __builtin_amdgcn_wmma_f32_16x16x32_bf16(false, av, false, b.v,
                                                  (short)0, acc, false, false);
  }

  // store: rows mt*16.. are within one t (64 rows per t, 16-row tiles)
  int t    = mt >> 2;
  int brow = mt & 3;
  size_t base = (((size_t)t * 4 + gate) * BSZ) * HID;
  #pragma unroll
  for (int r = 0; r < 8; ++r) {
    int m = r + half * 8;
    G[base + (size_t)(brow * 16 + m) * HID + nt * 16 + l15] = acc[r];
  }
}

// ---- grid barrier: monotonic counter, deterministic per launch -------------
__device__ __forceinline__ void grid_sync(unsigned* cnt, unsigned* gen, int step) {
  __syncthreads();
  if (threadIdx.x == 0) {
    __threadfence();
    unsigned prev = atomicAdd(cnt, 1u);
    unsigned target = (unsigned)(step + 1) * NBLK_REC;
    if (prev + 1 == target) {
      __hip_atomic_store(gen, (unsigned)(step + 1), __ATOMIC_RELEASE, __HIP_MEMORY_SCOPE_AGENT);
    } else {
      while (__hip_atomic_load(gen, __ATOMIC_ACQUIRE, __HIP_MEMORY_SCOPE_AGENT) <
             (unsigned)(step + 1)) {
        __builtin_amdgcn_s_sleep(2);
      }
    }
    __threadfence();
  }
  __syncthreads();
}

// ---- Phase 3: persistent recurrent kernel ----------------------------------
__global__ __launch_bounds__(128, 1) void lstm_rec(
    const float* __restrict__ G,                 // [T][4][B][H]
    const __hip_bfloat16* __restrict__ wbf,      // [4][H][DH], Wh = cols [256,512)
    __hip_bfloat16* __restrict__ hbuf,           // [2][B][H]
    unsigned* __restrict__ bar,
    float* __restrict__ out) {                   // stacked | hx | cx
  const int lane = threadIdx.x & 31;
  const int w    = blockIdx.x * 4 + (threadIdx.x >> 5);  // 0..63
  const int brow = w >> 4;                               // 0..3
  const int hcol = w & 15;                               // 0..15
  const int half = lane >> 4;
  const int l15  = lane & 15;

  // Cache this wave's Wh slice in VGPRs: 4 gates x 8 k-chunks (256 VGPRs).
  FragB16 bfr[4][8];
  #pragma unroll
  for (int g2 = 0; g2 < 4; ++g2) {
    const __hip_bfloat16* wrow =
        wbf + ((size_t)g2 * HID + (size_t)(hcol * 16 + l15)) * DH + DIN;
    #pragma unroll
    for (int kc = 0; kc < 8; ++kc) {
      int kb = kc * 32 + half * 16;
      bfr[g2][kc].q[0] = *(const uint4*)(wrow + kb);
      bfr[g2][kc].q[1] = *(const uint4*)(wrow + kb + 8);
    }
  }

  v8f c;
  #pragma unroll
  for (int r = 0; r < 8; ++r) c[r] = 0.0f;

  unsigned* cnt = bar;
  unsigned* gen = bar + 1;
  const int arow = brow * 16 + l15;
  const size_t SSTACK = (size_t)T_STEPS * BSZ * HID;

  // Per-lane element offsets within a [B][H] slab (m = r + half*8, n = l15)
  size_t eoff[8];
  #pragma unroll
  for (int r = 0; r < 8; ++r)
    eoff[r] = (size_t)(brow * 16 + r + half * 8) * HID + hcol * 16 + l15;

  // Software pipeline: G tile for step t held in registers across the barrier.
  // G is read-once -> non-temporal (preserve L2 for weights + h traffic).
  v8f gcur[4];
  #pragma unroll
  for (int g2 = 0; g2 < 4; ++g2)
    #pragma unroll
    for (int r = 0; r < 8; ++r)
      gcur[g2][r] = __builtin_nontemporal_load(G + (size_t)g2 * (BSZ * HID) + eoff[r]);

  for (int t = 0; t < T_STEPS; ++t) {
    // A fragments: previous h (bf16, double-buffered) -- only post-barrier load
    const __hip_bfloat16* hsrc =
        hbuf + (size_t)(t & 1) * (BSZ * HID) + (size_t)arow * HID;
    FragB16 afr[8];
    #pragma unroll
    for (int kc = 0; kc < 8; ++kc) {
      int klo = kc * 32 + half * 16;
      afr[kc].q[0] = *(const uint4*)(hsrc + klo);
      afr[kc].q[1] = *(const uint4*)(hsrc + klo + 8);
    }

    // 32 WMMAs: 4 independent chains of depth 8; pipelined G is the C operand
    v8f acc[4];
    #pragma unroll
    for (int g2 = 0; g2 < 4; ++g2) {
      acc[g2] = __builtin_amdgcn_wmma_f32_16x16x32_bf16(
          false, afr[0].v, false, bfr[g2][0].v, (short)0, gcur[g2], false, false);
      #pragma unroll
      for (int kc = 1; kc < 8; ++kc) {
        acc[g2] = __builtin_amdgcn_wmma_f32_16x16x32_bf16(
            false, afr[kc].v, false, bfr[g2][kc].v, (short)0, acc[g2], false, false);
      }
    }

    // Issue next step's G loads NOW: they retire during the tanh chain,
    // the stores, and the barrier spin (G is independent of the recurrence).
    if (t + 1 < T_STEPS) {
      size_t gnext = (size_t)(t + 1) * 4 * (BSZ * HID);
      #pragma unroll
      for (int g2 = 0; g2 < 4; ++g2)
        #pragma unroll
        for (int r = 0; r < 8; ++r)
          gcur[g2][r] = __builtin_nontemporal_load(
              G + gnext + (size_t)g2 * (BSZ * HID) + eoff[r]);
    }

    // Elementwise LSTM cell update, all in registers
    __hip_bfloat16* hdst = hbuf + (size_t)((t + 1) & 1) * (BSZ * HID);
    #pragma unroll
    for (int r = 0; r < 8; ++r) {
      float f = sigm_hw(acc[0][r]);
      float i = sigm_hw(acc[1][r]);
      float g = tanh_hw(acc[2][r]);
      float o = sigm_hw(acc[3][r]);
      float cn = __builtin_fmaf(f, c[r], i * g);
      c[r] = cn;
      float h = o * tanh_hw(cn);
      // stacked output is write-once: NT store, don't pollute L2
      __builtin_nontemporal_store(h, out + (size_t)t * (BSZ * HID) + eoff[r]);
      hdst[eoff[r]] = __float2bfloat16(h);      // next step's A operand: keep cached
      if (t == T_STEPS - 1) {
        out[SSTACK + eoff[r]] = h;              // hx
        out[SSTACK + BSZ * HID + eoff[r]] = cn; // cx
      }
    }

    // Fast arrive/wait if the runtime dispatched these 16 WGs as a cluster
    // (architecturally a NOP when ClusterID==0); the atomic protocol below
    // remains the general-case ordering + visibility guarantee.
    __builtin_amdgcn_s_cluster_barrier();
    grid_sync(cnt, gen, t);
  }
}

// ---------------------------------------------------------------------------
extern "C" void kernel_launch(void* const* d_in, const int* in_sizes, int n_in,
                              void* d_out, int out_size, void* d_ws, size_t ws_size,
                              hipStream_t stream) {
  (void)in_sizes; (void)n_in; (void)out_size; (void)ws_size;
  const float* x   = (const float*)d_in[0];
  const float* Wf  = (const float*)d_in[1];
  const float* bf_ = (const float*)d_in[2];
  const float* Wi  = (const float*)d_in[3];
  const float* bi_ = (const float*)d_in[4];
  const float* Wg  = (const float*)d_in[5];
  const float* bg_ = (const float*)d_in[6];
  const float* Wo  = (const float*)d_in[7];
  const float* bo_ = (const float*)d_in[8];

  char* ws = (char*)d_ws;
  __hip_bfloat16* wbf  = (__hip_bfloat16*)(ws + WBF_OFF);
  float*          G    = (float*)(ws + G_OFF);
  __hip_bfloat16* hbuf = (__hip_bfloat16*)(ws + HBUF_OFF);
  unsigned*       bar  = (unsigned*)(ws + BAR_OFF);
  float*          out  = (float*)d_out;

  cvt_weights<<<(4 * HID * DH + 255) / 256, 256, 0, stream>>>(Wf, Wi, Wg, Wo, wbf);
  init_state<<<(2 * BSZ * HID + 255) / 256, 256, 0, stream>>>(hbuf, bar);
  // 4 gates * 4096 M-tiles * 16 N-tiles = 262144 tiles, 8 waves/block
  xproj<<<262144 / 8, 256, 0, stream>>>(x, wbf, bf_, bi_, bg_, bo_, G);
  lstm_rec<<<NBLK_REC, 128, 0, stream>>>(G, wbf, hbuf, bar, out);
}
